// SHoMRFusion_74715251081763
// MI455X (gfx1250) — compile-verified, matrix operand order checked by reading, not run on hardware
//
#include <hip/hip_runtime.h>
#include <cstdint>
#include <cstddef>

#define DEV __device__ __forceinline__

typedef __attribute__((ext_vector_type(16))) __bf16 v16bf;
typedef __attribute__((ext_vector_type(8)))  float  v8f;
typedef __attribute__((ext_vector_type(2)))  float  v2f;

union FragBF { v16bf v; uint4 u[2]; };
union Chunk8 { uint4 u; __bf16 h[8]; };
union V2U    { v2f v; float2 f; };

static constexpr int Bn = 16384;
static constexpr int D  = 1024;

DEV float gelu_f(float x)   { return 0.5f * x * (1.0f + erff(x * 0.7071067811865476f)); }
DEV float sigmoid_f(float x){ return 1.0f / (1.0f + expf(-x)); }

// ---------------------------------------------------------------------------
// Weight packing: fragment-major layouts so GEMM B-operand loads are coalesced
// bf16 pack: chunk index = (((ntile*(K/32)+kb)*32 + lane)*2 + c)*8 + e
//   n = ntile*16 + (lane&15); k = kb*32 + (lane>=16 ? 8 : 0) + c*16 + e
// ---------------------------------------------------------------------------
__global__ void k_pack_bf16(const float* __restrict__ W, int K, int N,
                            __bf16* __restrict__ out) {
  size_t idx = (size_t)blockIdx.x * blockDim.x + threadIdx.x;
  size_t nchunks = ((size_t)K * N) >> 3;
  if (idx >= nchunks) return;
  int c    = (int)(idx & 1);
  int lane = (int)((idx >> 1) & 31);
  size_t t = idx >> 6;
  int kblocks = K >> 5;
  int kb    = (int)(t % kblocks);
  int ntile = (int)(t / kblocks);
  int n     = ntile * 16 + (lane & 15);
  int kbase = kb * 32 + ((lane >> 4) ? 8 : 0) + c * 16;
  Chunk8 ch;
#pragma unroll
  for (int e = 0; e < 8; ++e) ch.h[e] = (__bf16)W[(size_t)(kbase + e) * N + n];
  *reinterpret_cast<uint4*>(out + (idx << 3)) = ch.u;
}

// f32 pack (for the f32 WMMA path): chunk per (ntile, kb4, lane), 2 floats
//   n = ntile*16 + (lane&15); k = kb4*4 + (lane>=16 ? 2 : 0) + e
__global__ void k_pack_f32(const float* __restrict__ W, int K, int N,
                           float* __restrict__ out) {
  size_t idx = (size_t)blockIdx.x * blockDim.x + threadIdx.x;
  size_t nchunks = ((size_t)K * N) >> 1;
  if (idx >= nchunks) return;
  int lane = (int)(idx & 31);
  size_t t = idx >> 5;
  int kb4s = K >> 2;
  int kb4   = (int)(t % kb4s);
  int ntile = (int)(t / kb4s);
  int n  = ntile * 16 + (lane & 15);
  int kb = kb4 * 4 + ((lane >> 4) ? 2 : 0);
  float2 v;
  v.x = W[(size_t)kb * N + n];
  v.y = W[(size_t)(kb + 1) * N + n];
  *reinterpret_cast<float2*>(out + (idx << 1)) = v;
}

__global__ void k_cvt(const float* __restrict__ x, __bf16* __restrict__ y, size_t n) {
  size_t i = ((size_t)blockIdx.x * blockDim.x + threadIdx.x) * 4;
  if (i >= n) return;
  float4 f = *reinterpret_cast<const float4*>(x + i);
  y[i + 0] = (__bf16)f.x;
  y[i + 1] = (__bf16)f.y;
  y[i + 2] = (__bf16)f.z;
  y[i + 3] = (__bf16)f.w;
}

// ---------------------------------------------------------------------------
// bf16 GEMM core: WG tile 64(M)x128(N), 8 waves (2M x 4N), 2x2 subtiles/wave.
// Double-buffered LDS A-staging (one barrier per K-block), B prefetch.
// ---------------------------------------------------------------------------
DEV void gemm_core_bf16(const __bf16* __restrict__ A0, const __bf16* __restrict__ A1,
                        int lda, int ksplit, int K,
                        const __bf16* __restrict__ Bp,
                        int row_base, int ntile_base,
                        __bf16* lds, v8f acc[2][2]) {
  const int tid = threadIdx.x;
  const int lane = tid & 31, wave = tid >> 5;
  const int wm = wave >> 2, wn = wave & 3;
  const int lhalf = lane >> 4, lrow = lane & 15;
#pragma unroll
  for (int s = 0; s < 2; ++s)
#pragma unroll
    for (int t = 0; t < 2; ++t)
#pragma unroll
      for (int i = 0; i < 8; ++i) acc[s][t][i] = 0.0f;

  const int kblocks = K >> 5;
  const int srow = tid >> 2, sseg = tid & 3;
  const int LSTR = 40;
  const int LBUF = 64 * LSTR;

  auto stage = [&](int kb, int buf) {
    const int k0 = kb << 5;
    const __bf16* src = (k0 < ksplit)
        ? A0 + (size_t)(row_base + srow) * lda + k0
        : A1 + (size_t)(row_base + srow) * lda + (k0 - ksplit);
    *reinterpret_cast<uint4*>(&lds[buf * LBUF + srow * LSTR + sseg * 8]) =
        *reinterpret_cast<const uint4*>(src + sseg * 8);
  };
  stage(0, 0);
  for (int kb = 0; kb < kblocks; ++kb) {
    __syncthreads();
    if (kb + 1 < kblocks) stage(kb + 1, (kb + 1) & 1);
    const __bf16* lb = &lds[(kb & 1) * LBUF];
    FragBF afr[2];
#pragma unroll
    for (int s = 0; s < 2; ++s) {
      const __bf16* p = &lb[(wm * 32 + s * 16 + lrow) * LSTR + lhalf * 8];
      afr[s].u[0] = *reinterpret_cast<const uint4*>(p);
      afr[s].u[1] = *reinterpret_cast<const uint4*>(p + 16);
    }
    FragBF bfr[2];
#pragma unroll
    for (int t = 0; t < 2; ++t) {
      const int ntile = ntile_base + wn * 2 + t;
      const __bf16* p = Bp + (((size_t)ntile * kblocks + kb) << 9) + lane * 16;
      bfr[t].u[0] = *reinterpret_cast<const uint4*>(p);
      bfr[t].u[1] = *reinterpret_cast<const uint4*>(p + 8);
      if (kb + 1 < kblocks) __builtin_prefetch(p + 512, 0, 1);  // next K-block chunk
    }
#pragma unroll
    for (int s = 0; s < 2; ++s)
#pragma unroll
      for (int t = 0; t < 2; ++t)
        acc[s][t] = __builtin_amdgcn_wmma_f32_16x16x32_bf16(
            false, afr[s].v, false, bfr[t].v, (short)0, acc[s][t], false, false);
  }
}

// f32 GEMM core (V_WMMA_F32_16X16X4_F32) for precision-critical conf/route.
DEV void gemm_core_f32(const float* __restrict__ A0, const float* __restrict__ A1,
                       int lda, int ksplit, int K,
                       const float* __restrict__ Bp,
                       int row_base, int ntile_base,
                       float* lds, v8f acc[2][2]) {
  const int tid = threadIdx.x;
  const int lane = tid & 31, wave = tid >> 5;
  const int wm = wave >> 2, wn = wave & 3;
  const int lhalf = lane >> 4, lrow = lane & 15;
#pragma unroll
  for (int s = 0; s < 2; ++s)
#pragma unroll
    for (int t = 0; t < 2; ++t)
#pragma unroll
      for (int i = 0; i < 8; ++i) acc[s][t][i] = 0.0f;

  const int kb4s = K >> 2;
  const int kblocks = K >> 5;
  const int srow = tid >> 2, sseg = tid & 3;
  const int LSTR = 36;
  const int LBUF = 64 * LSTR;

  auto stage = [&](int kb, int buf) {
    const int k0 = kb << 5;
    const float* src = (k0 < ksplit)
        ? A0 + (size_t)(row_base + srow) * lda + k0
        : A1 + (size_t)(row_base + srow) * lda + (k0 - ksplit);
    float* dst = &lds[buf * LBUF + srow * LSTR + sseg * 8];
    *reinterpret_cast<float4*>(dst)     = *reinterpret_cast<const float4*>(src + sseg * 8);
    *reinterpret_cast<float4*>(dst + 4) = *reinterpret_cast<const float4*>(src + sseg * 8 + 4);
  };
  stage(0, 0);
  for (int kb = 0; kb < kblocks; ++kb) {
    __syncthreads();
    if (kb + 1 < kblocks) stage(kb + 1, (kb + 1) & 1);
    const float* lb = &lds[(kb & 1) * LBUF];
    const int k0 = kb << 5;
#pragma unroll
    for (int ks = 0; ks < 8; ++ks) {
      const int kk = ks * 4;
      V2U av[2];
#pragma unroll
      for (int s = 0; s < 2; ++s)
        av[s].f = *reinterpret_cast<const float2*>(
            &lb[(wm * 32 + s * 16 + lrow) * LSTR + kk + lhalf * 2]);
      V2U bv[2];
#pragma unroll
      for (int t = 0; t < 2; ++t) {
        const int ntile = ntile_base + wn * 2 + t;
        const float* p = Bp + (((size_t)ntile * kb4s + (k0 >> 2) + ks) * 32 + lane) * 2;
        bv[t].f = *reinterpret_cast<const float2*>(p);
        if (ks == 0 && kb + 1 < kblocks) __builtin_prefetch(p + 512, 0, 1);  // next K-block
      }
#pragma unroll
      for (int s = 0; s < 2; ++s)
#pragma unroll
        for (int t = 0; t < 2; ++t)
          acc[s][t] = __builtin_amdgcn_wmma_f32_16x16x4_f32(
              false, av[s].v, false, bv[t].v, (short)0, acc[s][t], false, false);
    }
  }
}

// ---------------------------------------------------------------------------
// conf/route projections (f32 WMMA) with fused tiny second matmul -> partials
// ---------------------------------------------------------------------------
__global__ void __launch_bounds__(256)
k_cr32(const float* __restrict__ img, const float* __restrict__ txt,
       const float* __restrict__ confp, const float* __restrict__ routep,
       const float* __restrict__ conf_b1, const float* __restrict__ route_b1,
       const float* __restrict__ conf_w2, const float* __restrict__ route_w2,
       float* __restrict__ Pc, float* __restrict__ Pr) {
  __shared__ float lds[2 * 64 * 36];
  const int z = blockIdx.z;
  const float* Bp   = z ? routep   : confp;
  const float* bias = z ? route_b1 : conf_b1;
  const float* W2   = z ? route_w2 : conf_w2;
  float* Pp = z ? Pr : Pc;
  const int J = z ? 3 : 2;
  const int row_base = blockIdx.x * 64, ntile_base = blockIdx.y * 8;
  v8f acc[2][2];
  gemm_core_f32(img, txt, D, D, 2 * D, Bp, row_base, ntile_base, lds, acc);

  const int lane = threadIdx.x & 31, lhalf = lane >> 4, lrow = lane & 15;
  const int wave = threadIdx.x >> 5, wm = wave >> 2, wn = wave & 3;
  float part[2][8][3];
#pragma unroll
  for (int s = 0; s < 2; ++s)
#pragma unroll
    for (int i = 0; i < 8; ++i)
#pragma unroll
      for (int j = 0; j < 3; ++j) part[s][i][j] = 0.0f;

#pragma unroll
  for (int s = 0; s < 2; ++s)
#pragma unroll
    for (int t = 0; t < 2; ++t)
#pragma unroll
      for (int i = 0; i < 8; ++i) {
        int n = (ntile_base + wn * 2 + t) * 16 + lrow;
        float h = gelu_f(acc[s][t][i] + bias[n]);
#pragma unroll
        for (int j = 0; j < 3; ++j)
          if (j < J) part[s][i][j] += h * W2[n * J + j];
      }
  const int pidx = blockIdx.y * 4 + wn;  // 0..31 deterministic partial slot
#pragma unroll
  for (int s = 0; s < 2; ++s)
#pragma unroll
    for (int i = 0; i < 8; ++i)
#pragma unroll
      for (int j = 0; j < 3; ++j) {
        if (j >= J) continue;
        float v = part[s][i][j];
        v += __shfl_xor(v, 8, 32);
        v += __shfl_xor(v, 4, 32);
        v += __shfl_xor(v, 2, 32);
        v += __shfl_xor(v, 1, 32);
        if (lrow == 0) {
          int m = row_base + wm * 32 + s * 16 + lhalf * 8 + i;
          Pp[((size_t)pidx * Bn + m) * J + j] = v;
        }
      }
}

__global__ void k_logits(const float* __restrict__ Pc, const float* __restrict__ Pr,
                         const float* __restrict__ cb2, const float* __restrict__ rb2,
                         float* __restrict__ confL, float* __restrict__ routeL) {
  int row = blockIdx.x * blockDim.x + threadIdx.x;
  if (row >= Bn) return;
  float c0 = cb2[0], c1 = cb2[1];
  float r0 = rb2[0], r1 = rb2[1], r2 = rb2[2];
  for (int p = 0; p < 32; ++p) {
    c0 += Pc[((size_t)p * Bn + row) * 2 + 0];
    c1 += Pc[((size_t)p * Bn + row) * 2 + 1];
    r0 += Pr[((size_t)p * Bn + row) * 3 + 0];
    r1 += Pr[((size_t)p * Bn + row) * 3 + 1];
    r2 += Pr[((size_t)p * Bn + row) * 3 + 2];
  }
  confL[row * 2 + 0] = c0; confL[row * 2 + 1] = c1;
  routeL[row * 3 + 0] = r0; routeL[row * 3 + 1] = r1; routeL[row * 3 + 2] = r2;
}

// ---------------------------------------------------------------------------
// Generic bf16 GEMMs. act: 0=none 1=gelu 2=sigmoid
// ---------------------------------------------------------------------------
__global__ void __launch_bounds__(256)
k_gemm_bf16(const __bf16* __restrict__ A0, const __bf16* __restrict__ A1,
            int ksplit, int lda,
            const __bf16* __restrict__ Bp, const float* __restrict__ bias,
            __bf16* __restrict__ out, int K, int N, int act) {
  __shared__ __bf16 lds[2 * 64 * 40];
  const int row_base = blockIdx.x * 64, ntile_base = blockIdx.y * 8;
  v8f acc[2][2];
  gemm_core_bf16(A0, A1, lda, ksplit, K, Bp, row_base, ntile_base, lds, acc);
  const int lane = threadIdx.x & 31, lhalf = lane >> 4, lrow = lane & 15;
  const int wave = threadIdx.x >> 5, wm = wave >> 2, wn = wave & 3;
#pragma unroll
  for (int s = 0; s < 2; ++s)
#pragma unroll
    for (int t = 0; t < 2; ++t)
#pragma unroll
      for (int i = 0; i < 8; ++i) {
        int m = row_base + wm * 32 + s * 16 + lhalf * 8 + i;
        int n = (ntile_base + wn * 2 + t) * 16 + lrow;
        float v = acc[s][t][i] + bias[n];
        if (act == 1) v = gelu_f(v);
        else if (act == 2) v = sigmoid_f(v);
        out[(size_t)m * N + n] = (__bf16)v;
      }
}

__global__ void __launch_bounds__(256)
k_gemm_f32out(const __bf16* __restrict__ A0, const __bf16* __restrict__ A1,
              int ksplit, int lda,
              const __bf16* __restrict__ Bp, const float* __restrict__ bias,
              const float* __restrict__ res, float* __restrict__ out,
              int K, int N, int act) {
  __shared__ __bf16 lds[2 * 64 * 40];
  const int row_base = blockIdx.x * 64, ntile_base = blockIdx.y * 8;
  v8f acc[2][2];
  gemm_core_bf16(A0, A1, lda, ksplit, K, Bp, row_base, ntile_base, lds, acc);
  const int lane = threadIdx.x & 31, lhalf = lane >> 4, lrow = lane & 15;
  const int wave = threadIdx.x >> 5, wm = wave >> 2, wn = wave & 3;
#pragma unroll
  for (int s = 0; s < 2; ++s)
#pragma unroll
    for (int t = 0; t < 2; ++t)
#pragma unroll
      for (int i = 0; i < 8; ++i) {
        int m = row_base + wm * 32 + s * 16 + lhalf * 8 + i;
        int n = (ntile_base + wn * 2 + t) * 16 + lrow;
        float v = acc[s][t][i] + bias[n];
        if (act == 1) v = gelu_f(v);
        if (res) v += res[(size_t)m * N + n];
        out[(size_t)m * N + n] = v;
      }
}

// ---------------------------------------------------------------------------
// S=2 attention, folded: o_mean = 0.5*((a_ii+a_ti) v_i + (a_it+a_tt) v_t)
// ---------------------------------------------------------------------------
__global__ void __launch_bounds__(256)
k_attn(const __bf16* __restrict__ qib, const __bf16* __restrict__ qtb,
       __bf16* __restrict__ omean) {
  const int t = threadIdx.x;
  const int row = blockIdx.x * 32 + (t >> 3);
  const int head = t & 7;
  const __bf16* bi = qib + (size_t)row * 3072 + head * 128;
  const __bf16* bt = qtb + (size_t)row * 3072 + head * 128;
  float sii = 0, sit = 0, sti = 0, stt = 0;
  for (int d = 0; d < 128; d += 8) {
    Chunk8 qi, ki, qt, kt;
    qi.u = *reinterpret_cast<const uint4*>(bi + d);
    ki.u = *reinterpret_cast<const uint4*>(bi + 1024 + d);
    qt.u = *reinterpret_cast<const uint4*>(bt + d);
    kt.u = *reinterpret_cast<const uint4*>(bt + 1024 + d);
#pragma unroll
    for (int e = 0; e < 8; ++e) {
      float a = (float)qi.h[e], b = (float)ki.h[e];
      float c = (float)qt.h[e], dd = (float)kt.h[e];
      sii += a * b; sit += a * dd; sti += c * b; stt += c * dd;
    }
  }
  const float scale = 0.08838834764831845f;  // 1/sqrt(128)
  float li0 = sii * scale, li1 = sit * scale;
  float mi = fmaxf(li0, li1);
  float ei0 = expf(li0 - mi), ei1 = expf(li1 - mi);
  float isi = 1.0f / (ei0 + ei1);
  float aii = ei0 * isi, ait = ei1 * isi;
  float lt0 = sti * scale, lt1 = stt * scale;
  float mt = fmaxf(lt0, lt1);
  float et0 = expf(lt0 - mt), et1 = expf(lt1 - mt);
  float ist = 1.0f / (et0 + et1);
  float ati = et0 * ist, att = et1 * ist;
  float ci = 0.5f * (aii + ati), ct = 0.5f * (ait + att);
  __bf16* o = omean + (size_t)row * 1024 + head * 128;
  for (int d = 0; d < 128; d += 8) {
    Chunk8 vi, vt, ov;
    vi.u = *reinterpret_cast<const uint4*>(bi + 2048 + d);
    vt.u = *reinterpret_cast<const uint4*>(bt + 2048 + d);
#pragma unroll
    for (int e = 0; e < 8; ++e)
      ov.h[e] = (__bf16)(ci * (float)vi.h[e] + ct * (float)vt.h[e]);
    *reinterpret_cast<uint4*>(o + d) = ov.u;
  }
}

// ---------------------------------------------------------------------------
// Combine (soft/both pre-activations) + LayerNorm + routing flags
// ---------------------------------------------------------------------------
__global__ void __launch_bounds__(256)
k_combine(const float* __restrict__ img, const float* __restrict__ txt,
          const float* __restrict__ confL, const float* __restrict__ routeL,
          const __bf16* __restrict__ gateb, const float* __restrict__ attnm,
          const float* __restrict__ g1, const float* __restrict__ be1,
          const float* __restrict__ g2, const float* __restrict__ be2,
          __bf16* __restrict__ softb, float* __restrict__ softr,
          __bf16* __restrict__ bothb, float* __restrict__ bothr,
          int* __restrict__ flags) {
  __shared__ float red[256];
  const int tid = threadIdx.x;
  const int rloc = tid >> 7, lt = tid & 127;
  const int row = blockIdx.x * 2 + rloc;

  float c0 = confL[row * 2 + 0], c1 = confL[row * 2 + 1];
  float cm = fmaxf(c0, c1);
  float e0 = expf(c0 - cm), e1 = expf(c1 - cm);
  float invs = 1.0f / (e0 + e1);
  float wv = e0 * invs, wt = e1 * invs;
  float maxconf = fmaxf(wv, wt);
  int use_hard = maxconf > 0.6f;
  int low = (maxconf < 0.3f) && !use_hard;
  float r0 = routeL[row * 3 + 0], r1 = routeL[row * 3 + 1], r2 = routeL[row * 3 + 2];
  int route = 0; float best = r0;
  if (r1 > best) { best = r1; route = 1; }
  if (r2 > best) { best = r2; route = 2; }
  if (lt == 0) flags[row] = route | (use_hard << 2) | (low << 3);

  float sp[8], bp[8];
  const size_t base = (size_t)row * D;
#pragma unroll
  for (int e = 0; e < 8; ++e) {
    int col = lt * 8 + e;
    float iv = img[base + col], tv = txt[base + col];
    float am = attnm[base + col];
    float g = (float)gateb[base + col];
    sp[e] = wv * iv + wt * tv + am;
    bp[e] = g * iv + (1.0f - g) * tv + am;
  }
  auto reduce = [&](float v) -> float {
    red[tid] = v; __syncthreads();
    for (int off = 64; off > 0; off >>= 1) {
      if (lt < off) red[rloc * 128 + lt] += red[rloc * 128 + lt + off];
      __syncthreads();
    }
    float r = red[rloc * 128]; __syncthreads();
    return r;
  };
  float ssum = 0, bsum = 0;
#pragma unroll
  for (int e = 0; e < 8; ++e) { ssum += sp[e]; bsum += bp[e]; }
  float smean = reduce(ssum) * (1.0f / D);
  float bmean = reduce(bsum) * (1.0f / D);
  float s2 = 0, b2 = 0;
#pragma unroll
  for (int e = 0; e < 8; ++e) {
    float d1 = sp[e] - smean; s2 += d1 * d1;
    float d2 = bp[e] - bmean; b2 += d2 * d2;
  }
  float svar = reduce(s2) * (1.0f / D);
  float bvar = reduce(b2) * (1.0f / D);
  float sinv = rsqrtf(svar + 1e-5f), binv = rsqrtf(bvar + 1e-5f);
#pragma unroll
  for (int e = 0; e < 8; ++e) {
    int col = lt * 8 + e;
    float xs = (sp[e] - smean) * sinv * g1[col] + be1[col];
    float xb = (bp[e] - bmean) * binv * g2[col] + be2[col];
    softr[base + col] = xs; softb[base + col] = (__bf16)xs;
    bothr[base + col] = xb; bothb[base + col] = (__bf16)xb;
  }
}

__global__ void k_final(const float* __restrict__ img, const float* __restrict__ txt,
                        const float* __restrict__ softo, const float* __restrict__ botho,
                        const float* __restrict__ visf, const float* __restrict__ texf,
                        const int* __restrict__ flags, float* __restrict__ out) {
  size_t idx = (size_t)blockIdx.x * blockDim.x + threadIdx.x;
  if (idx >= (size_t)Bn * D) return;
  size_t row = idx >> 10;
  int f = flags[row];
  int route = f & 3;
  bool uh = (f & 4) != 0, low = (f & 8) != 0;
  float v;
  if (low)      v = 0.5f * (img[idx] + txt[idx]);
  else if (uh)  v = (route == 0) ? visf[idx] : (route == 1) ? texf[idx] : botho[idx];
  else          v = softo[idx];
  out[idx] = v;
}

// ---------------------------------------------------------------------------
extern "C" void kernel_launch(void* const* d_in, const int* in_sizes, int n_in,
                              void* d_out, int out_size, void* d_ws, size_t ws_size,
                              hipStream_t stream) {
  (void)in_sizes; (void)n_in; (void)out_size; (void)ws_size;
  const float* img      = (const float*)d_in[0];
  const float* txt      = (const float*)d_in[1];
  const float* conf_w1  = (const float*)d_in[2];
  const float* conf_b1  = (const float*)d_in[3];
  const float* conf_w2  = (const float*)d_in[4];
  const float* conf_b2  = (const float*)d_in[5];
  const float* route_w1 = (const float*)d_in[6];
  const float* route_b1 = (const float*)d_in[7];
  const float* route_w2 = (const float*)d_in[8];
  const float* route_b2 = (const float*)d_in[9];
  const float* wqkv     = (const float*)d_in[10];
  const float* bqkv     = (const float*)d_in[11];
  const float* wo       = (const float*)d_in[12];
  const float* bo       = (const float*)d_in[13];
  const float* gate_w   = (const float*)d_in[14];
  const float* gate_b   = (const float*)d_in[15];
  const float* g1       = (const float*)d_in[16];
  const float* be1      = (const float*)d_in[17];
  const float* g2       = (const float*)d_in[18];
  const float* be2      = (const float*)d_in[19];
  const float* ffn_w1   = (const float*)d_in[20];
  const float* ffn_b1   = (const float*)d_in[21];
  const float* ffn_w2   = (const float*)d_in[22];
  const float* ffn_b2   = (const float*)d_in[23];
  const float* proj_w   = (const float*)d_in[24];
  const float* proj_b   = (const float*)d_in[25];

  char* ws = (char*)d_ws;
  size_t off = 0;
  auto alloc = [&](size_t bytes) -> char* {
    char* p = ws + off;
    off += (bytes + 255) & ~(size_t)255;
    return p;
  };
  __bf16* imgb   = (__bf16*)alloc((size_t)Bn * D * 2);
  __bf16* txtb   = (__bf16*)alloc((size_t)Bn * D * 2);
  float*  confp  = (float*)alloc((size_t)2048 * 1024 * 4);
  float*  routep = (float*)alloc((size_t)2048 * 1024 * 4);
  __bf16* gatep  = (__bf16*)alloc((size_t)2048 * 1024 * 2);
  __bf16* wqkvp  = (__bf16*)alloc((size_t)1024 * 3072 * 2);
  __bf16* wop    = (__bf16*)alloc((size_t)1024 * 1024 * 2);
  __bf16* projp  = (__bf16*)alloc((size_t)1024 * 1024 * 2);
  __bf16* ffn1p  = (__bf16*)alloc((size_t)1024 * 4096 * 2);
  __bf16* ffn2p  = (__bf16*)alloc((size_t)4096 * 1024 * 2);
  float*  Pc     = (float*)alloc((size_t)32 * Bn * 2 * 4);
  float*  Pr     = (float*)alloc((size_t)32 * Bn * 3 * 4);
  float*  confL  = (float*)alloc((size_t)Bn * 2 * 4);
  float*  routeL = (float*)alloc((size_t)Bn * 3 * 4);
  __bf16* gateb  = (__bf16*)alloc((size_t)Bn * D * 2);
  __bf16* qib    = (__bf16*)alloc((size_t)Bn * 3072 * 2);
  __bf16* qtb    = (__bf16*)alloc((size_t)Bn * 3072 * 2);
  __bf16* omeanb = (__bf16*)alloc((size_t)Bn * D * 2);
  float*  attnm  = (float*)alloc((size_t)Bn * D * 4);
  __bf16* softb  = (__bf16*)alloc((size_t)Bn * D * 2);
  __bf16* bothb  = (__bf16*)alloc((size_t)Bn * D * 2);
  float*  softr  = (float*)alloc((size_t)Bn * D * 4);
  float*  bothr  = (float*)alloc((size_t)Bn * D * 4);
  __bf16* h1b    = (__bf16*)alloc((size_t)Bn * 4096 * 2);
  float*  softo  = (float*)alloc((size_t)Bn * D * 4);
  float*  botho  = (float*)alloc((size_t)Bn * D * 4);
  float*  visf   = (float*)alloc((size_t)Bn * D * 4);
  float*  texf   = (float*)alloc((size_t)Bn * D * 4);
  int*    flags  = (int*)alloc((size_t)Bn * 4);

  auto packs_bf16 = [&](const float* W, int K, int N, __bf16* out) {
    size_t chunks = ((size_t)K * N) >> 3;
    k_pack_bf16<<<dim3((unsigned)((chunks + 255) / 256)), 256, 0, stream>>>(W, K, N, out);
  };
  auto packs_f32 = [&](const float* W, int K, int N, float* out) {
    size_t chunks = ((size_t)K * N) >> 1;
    k_pack_f32<<<dim3((unsigned)((chunks + 255) / 256)), 256, 0, stream>>>(W, K, N, out);
  };
  packs_f32(conf_w1, 2048, 1024, confp);
  packs_f32(route_w1, 2048, 1024, routep);
  packs_bf16(gate_w, 2048, 1024, gatep);
  packs_bf16(wqkv, 1024, 3072, wqkvp);
  packs_bf16(wo, 1024, 1024, wop);
  packs_bf16(proj_w, 1024, 1024, projp);
  packs_bf16(ffn_w1, 1024, 4096, ffn1p);
  packs_bf16(ffn_w2, 4096, 1024, ffn2p);

  size_t nel = (size_t)Bn * D;
  k_cvt<<<dim3((unsigned)((nel / 4 + 255) / 256)), 256, 0, stream>>>(img, imgb, nel);
  k_cvt<<<dim3((unsigned)((nel / 4 + 255) / 256)), 256, 0, stream>>>(txt, txtb, nel);

  const int MT = Bn / 64;  // 256 M-tiles

  // conf/route (f32 WMMA) + deterministic logit reduce
  k_cr32<<<dim3(MT, 8, 2), 256, 0, stream>>>(img, txt, confp, routep,
                                             conf_b1, route_b1, conf_w2, route_w2, Pc, Pr);
  k_logits<<<dim3(Bn / 256), 256, 0, stream>>>(Pc, Pr, conf_b2, route_b2, confL, routeL);

  // gate (bf16 WMMA, sigmoid)
  k_gemm_bf16<<<dim3(MT, 8), 256, 0, stream>>>(imgb, txtb, 1024, 1024, gatep, gate_b,
                                               gateb, 2048, 1024, 2);
  // qkv for both tokens
  k_gemm_bf16<<<dim3(MT, 24), 256, 0, stream>>>(imgb, imgb, 1024, 1024, wqkvp, bqkv,
                                                qib, 1024, 3072, 0);
  k_gemm_bf16<<<dim3(MT, 24), 256, 0, stream>>>(txtb, txtb, 1024, 1024, wqkvp, bqkv,
                                                qtb, 1024, 3072, 0);
  // attention (S=2 folded)
  k_attn<<<dim3(Bn / 32), 256, 0, stream>>>(qib, qtb, omeanb);
  // attn_mean = o_mean @ wo + bo
  k_gemm_f32out<<<dim3(MT, 8), 256, 0, stream>>>(omeanb, omeanb, 1024, 1024, wop, bo,
                                                 nullptr, attnm, 1024, 1024, 0);
  // combine + LN
  k_combine<<<dim3(Bn / 2), 256, 0, stream>>>(img, txt, confL, routeL, gateb, attnm,
                                              g1, be1, g2, be2,
                                              softb, softr, bothb, bothr, flags);
  // FFN (soft), h1b reused across passes
  k_gemm_bf16<<<dim3(MT, 32), 256, 0, stream>>>(softb, softb, 1024, 1024, ffn1p, ffn_b1,
                                                h1b, 1024, 4096, 1);
  k_gemm_f32out<<<dim3(MT, 8), 256, 0, stream>>>(h1b, h1b, 4096, 4096, ffn2p, ffn_b2,
                                                 softr, softo, 4096, 1024, 0);
  // FFN (both)
  k_gemm_bf16<<<dim3(MT, 32), 256, 0, stream>>>(bothb, bothb, 1024, 1024, ffn1p, ffn_b1,
                                                h1b, 1024, 4096, 1);
  k_gemm_f32out<<<dim3(MT, 8), 256, 0, stream>>>(h1b, h1b, 4096, 4096, ffn2p, ffn_b2,
                                                 bothr, botho, 4096, 1024, 0);
  // proj paths
  k_gemm_f32out<<<dim3(MT, 8), 256, 0, stream>>>(imgb, imgb, 1024, 1024, projp, proj_b,
                                                 nullptr, visf, 1024, 1024, 1);
  k_gemm_f32out<<<dim3(MT, 8), 256, 0, stream>>>(txtb, txtb, 1024, 1024, projp, proj_b,
                                                 nullptr, texf, 1024, 1024, 1);
  // final routing/select
  k_final<<<dim3((unsigned)((nel + 255) / 256)), 256, 0, stream>>>(
      img, txt, softo, botho, visf, texf, flags, (float*)d_out);
}
